// MLAAttention_70978629533878
// MI455X (gfx1250) — compile-verified, matrix operand order checked by reading, not run on hardware
//
#include <hip/hip_runtime.h>
#include <hip/hip_bf16.h>

// ---------------------------------------------------------------------------
// MLA attention for MI455X (gfx1250, wave32, WMMA + Tensor Data Mover).
// Pipeline:
//   gemm<0>: q      = x  @ wq^T        (f32 out, for RoPE)
//   gemm<0>: kvlat  = x  @ w_kv_down^T (f32 out)
//   gemm<0>: k      = kvlat @ w_k_up^T (f32 out, for RoPE)
//   gemm<1>: Vt     = (kvlat @ w_v_up^T)^T  (bf16, transposed [N,S] for PV frags)
//   rope   : qb,kb  = RoPE(q,k) -> bf16
//   flash  : att    = causal softmax(qb kb^T / 8) @ V   (online softmax, WMMA)
//   gemm<0>: out    = att @ wo^T
// GEMM A/B tiles are DMA'd into LDS by the TDM (tensor_load_to_lds), double
// buffered against the WMMA loop, tracked with TENSORcnt.
// ---------------------------------------------------------------------------

typedef __attribute__((ext_vector_type(16))) __bf16   v16bf;
typedef __attribute__((ext_vector_type(8)))  float    v8f;
typedef __attribute__((ext_vector_type(4)))  unsigned u32x4;
typedef __attribute__((ext_vector_type(8)))  unsigned u32x8;

#define SEQ     4096
#define DMODEL  1024
#define LATENT  512
#define NHEADS  16
#define HDIM    64
#define QDIM    1024   // NHEADS * HDIM

__device__ inline __bf16 f2bf(float f) {
  unsigned u = __builtin_bit_cast(unsigned, f);
  u = (u + 0x7FFFu + ((u >> 16) & 1u)) >> 16;              // round-to-nearest-even
  return __builtin_bit_cast(__bf16, (unsigned short)u);
}

// Pack two f32 into adjacent bf16 fragment slots (prefer v_cvt_pk_bf16_f32).
__device__ inline void pack2(v16bf& f, int v, float x, float y) {
#if __has_builtin(__builtin_amdgcn_cvt_pk_bf16_f32)
  typedef __attribute__((ext_vector_type(2))) __bf16 v2bf;
  v2bf p = __builtin_amdgcn_cvt_pk_bf16_f32(x, y);
  f[2 * v] = p[0];
  f[2 * v + 1] = p[1];
#else
  f[2 * v] = f2bf(x);
  f[2 * v + 1] = f2bf(y);
#endif
}

// Byte offset of a workgroup-shared object inside LDS (for the TDM D#).
typedef __attribute__((address_space(3))) const void* lds_cptr;
__device__ inline unsigned lds_addr_of(const void* p) {
  return (unsigned)(unsigned long long)(lds_cptr)p;
}

// TDM: DMA a 64-row x 32-col f32 tile (row stride = `stride` elems) from
// global `gp` into LDS at `lds_addr`.  2D descriptor, groups 2/3 disabled.
__device__ inline void tdm_load_tile(unsigned lds_addr, const float* gp,
                                     unsigned dim0, unsigned dim1,
                                     unsigned stride) {
  unsigned long long ga = (unsigned long long)gp;
  u32x4 g0;
  g0[0] = 1u;                                               // count=1 (valid)
  g0[1] = lds_addr;                                         // lds_addr
  g0[2] = (unsigned)ga;                                     // global_addr lo
  g0[3] = (unsigned)((ga >> 32) & 0x1FFFFFFu) | 0x80000000u; // ga hi | type=2
  u32x8 g1;
  g1[0] = 2u << 16;                                         // data_size = 4B
  g1[1] = (dim0 & 0xFFFFu) << 16;                           // tensor_dim0 lo
  g1[2] = (dim0 >> 16) | ((dim1 & 0xFFFFu) << 16);          // dim0 hi | dim1 lo
  g1[3] = (dim1 >> 16) | (32u << 16);                       // dim1 hi | tile_dim0=32
  g1[4] = 64u;                                              // tile_dim1=64, tile_dim2=0
  g1[5] = stride;                                           // tensor_dim0_stride lo
  g1[6] = 0u;
  g1[7] = 0u;
  asm volatile("tensor_load_to_lds %0, %1" :: "s"(g0), "s"(g1) : "memory");
}

// Build a 16x32 bf16 A-layout fragment from a f32 LDS tile row block.
// ISA 7.12.2 16-bit A layout: lane={m=lane&15, half=lane>>4},
// kk(v,half) = 2*(v&3) + (v>=4?16:0) + half*8.
__device__ inline v16bf frag_from_lds(const float t[][32], int row0) {
  int lane = threadIdx.x & 31;
  int m = lane & 15, half = lane >> 4;
  const float* rp = &t[row0 + m][half * 8];
  v16bf f;
#pragma unroll
  for (int v = 0; v < 8; ++v) {
    int kk = 2 * (v & 3) + ((v >> 2) << 4);
    pack2(f, v, rp[kk], rp[kk + 1]);
  }
  return f;
}

__device__ inline v16bf load_frag_bf16(const __bf16* __restrict__ p, int ld,
                                       int row0, int k0) {
  int lane = threadIdx.x & 31;
  int m = lane & 15, half = lane >> 4;
  const __bf16* rp = p + (size_t)(row0 + m) * ld + k0 + half * 8;
  v16bf f;
#pragma unroll
  for (int v = 0; v < 8; ++v) {
    int kk = 2 * (v & 3) + ((v >> 2) << 4);
    f[2 * v] = rp[kk];
    f[2 * v + 1] = rp[kk + 1];
  }
  return f;
}

// C = A[M,K] * Bw[N,K]^T.  Block = 4 waves = 64x64 tile; wave = 16x64 strip.
// TDM double-buffers 64x32 A and B tiles into LDS.
// MODE 0: f32 C[M,N].  MODE 1: bf16 transposed Ct[N,M].
template <int MODE>
__global__ __launch_bounds__(128) void gemm_xbt(const float* __restrict__ A,
                                                const float* __restrict__ Bw,
                                                void* __restrict__ Cout,
                                                int M, int N, int K) {
  __shared__ float tA[2][64][32];
  __shared__ float tB[2][64][32];
  int wave = threadIdx.x >> 5;
  int mblk = blockIdx.x * 64;
  int nbase = blockIdx.y * 64;
  v8f acc[4] = {};

  int nk = K >> 5;
  if (wave == 0) {
    tdm_load_tile(lds_addr_of(&tA[0][0][0]), A + (size_t)mblk * K, K, M, K);
    tdm_load_tile(lds_addr_of(&tB[0][0][0]), Bw + (size_t)nbase * K, K, N, K);
    __builtin_amdgcn_s_wait_tensorcnt(0);
  }
  __syncthreads();

  for (int t = 0; t < nk; ++t) {
    int cur = t & 1;
    if (wave == 0 && t + 1 < nk) {               // prefetch next tiles via TDM
      int k0 = (t + 1) * 32;
      tdm_load_tile(lds_addr_of(&tA[cur ^ 1][0][0]), A + (size_t)mblk * K + k0,
                    K, M, K);
      tdm_load_tile(lds_addr_of(&tB[cur ^ 1][0][0]), Bw + (size_t)nbase * K + k0,
                    K, N, K);
    }
    v16bf a = frag_from_lds(tA[cur], wave * 16);
#pragma unroll
    for (int nt = 0; nt < 4; ++nt) {
      v16bf b = frag_from_lds(tB[cur], nt * 16);
      acc[nt] = __builtin_amdgcn_wmma_f32_16x16x32_bf16(
          false, a, false, b, (short)0, acc[nt], false, false);
    }
    if (wave == 0) __builtin_amdgcn_s_wait_tensorcnt(0);
    __syncthreads();
  }

  int lane = threadIdx.x & 31;
  int col = lane & 15, half = lane >> 4;
#pragma unroll
  for (int nt = 0; nt < 4; ++nt) {
#pragma unroll
    for (int r = 0; r < 8; ++r) {
      int row = mblk + wave * 16 + r + half * 8;
      int cc = nbase + nt * 16 + col;
      if (MODE == 0)
        ((float*)Cout)[(size_t)row * N + cc] = acc[nt][r];
      else
        ((__bf16*)Cout)[(size_t)cc * M + row] = f2bf(acc[nt][r]);
    }
  }
}

// RoPE on q,k (f32 in) + convert to bf16.  One thread per (s, h, freq i<32).
__global__ void rope_cvt(const float* __restrict__ qf, const float* __restrict__ kf,
                         __bf16* __restrict__ qb, __bf16* __restrict__ kb) {
  int idx = blockIdx.x * blockDim.x + threadIdx.x;
  if (idx >= SEQ * NHEADS * 32) return;
  int i = idx & 31;
  int h = (idx >> 5) & (NHEADS - 1);
  int s = idx >> 9;
  // inv_freq = theta^(-2i/64) = exp(-i/32 * ln(10000))
  float inv = __expf(-(float)i * 0.28782313662425614f);
  float ang = (float)s * inv;
  float sn, c;
  __sincosf(ang, &sn, &c);
  size_t base = (size_t)s * QDIM + h * HDIM + i;
  float q1 = qf[base], q2 = qf[base + 32];
  float k1 = kf[base], k2 = kf[base + 32];
  qb[base]      = f2bf(q1 * c - q2 * sn);
  qb[base + 32] = f2bf(q2 * c + q1 * sn);
  kb[base]      = f2bf(k1 * c - k2 * sn);
  kb[base + 32] = f2bf(k2 * c + k1 * sn);
}

// Causal flash attention.  Block = 4 waves x 16 q-rows, grid = (S/64, H).
// qb,kb: bf16 [S, H*D]; vt: bf16 [H*D, S] (transposed); att: f32 [S, H*D].
__global__ __launch_bounds__(128) void mla_flash(const __bf16* __restrict__ qb,
                                                 const __bf16* __restrict__ kb,
                                                 const __bf16* __restrict__ vt,
                                                 float* __restrict__ att) {
  __shared__ __bf16 pbuf[4][16][32];   // per-wave P staging (C-layout -> A-layout)
  int wave = threadIdx.x >> 5;
  int lane = threadIdx.x & 31;
  int m = lane & 15, half = lane >> 4;
  int head = blockIdx.y;
  int qbase = blockIdx.x * 64 + wave * 16;

  v16bf aQ0 = load_frag_bf16(qb, QDIM, qbase, head * HDIM + 0);
  v16bf aQ1 = load_frag_bf16(qb, QDIM, qbase, head * HDIM + 32);

  v8f o[4] = {};
  float mrow[8], lrow[8];
#pragma unroll
  for (int r = 0; r < 8; ++r) { mrow[r] = -3.0e38f; lrow[r] = 0.f; }

  int qend = qbase + 15;
  for (int k0 = 0; k0 <= qend; k0 += 32) {
    // ---- scores: Q[16,64] x K^T -> two 16x16 tiles --------------------
    v8f sc[2];
#pragma unroll
    for (int j = 0; j < 2; ++j) {
      v16bf bK0 = load_frag_bf16(kb, QDIM, k0 + j * 16, head * HDIM + 0);
      v16bf bK1 = load_frag_bf16(kb, QDIM, k0 + j * 16, head * HDIM + 32);
      v8f c = {};
      c = __builtin_amdgcn_wmma_f32_16x16x32_bf16(false, aQ0, false, bK0,
                                                  (short)0, c, false, false);
      c = __builtin_amdgcn_wmma_f32_16x16x32_bf16(false, aQ1, false, bK1,
                                                  (short)0, c, false, false);
      sc[j] = c;
    }
    // ---- causal mask + scale + online softmax -------------------------
#pragma unroll
    for (int r = 0; r < 8; ++r) {
      int row = qbase + r + half * 8;
      int c0 = k0 + m, c1 = k0 + 16 + m;
      float v0 = (c0 > row) ? -3.0e38f : sc[0][r] * 0.125f;
      float v1 = (c1 > row) ? -3.0e38f : sc[1][r] * 0.125f;
      float mx = fmaxf(v0, v1);
      mx = fmaxf(mx, __shfl_xor(mx, 1, 32));
      mx = fmaxf(mx, __shfl_xor(mx, 2, 32));
      mx = fmaxf(mx, __shfl_xor(mx, 4, 32));
      mx = fmaxf(mx, __shfl_xor(mx, 8, 32));
      float mnew = fmaxf(mrow[r], mx);
      float p0 = __expf(v0 - mnew);
      float p1 = __expf(v1 - mnew);
      float alpha = __expf(mrow[r] - mnew);
      float ps = p0 + p1;
      ps += __shfl_xor(ps, 1, 32);
      ps += __shfl_xor(ps, 2, 32);
      ps += __shfl_xor(ps, 4, 32);
      ps += __shfl_xor(ps, 8, 32);
      lrow[r] = lrow[r] * alpha + ps;
      mrow[r] = mnew;
#pragma unroll
      for (int nt = 0; nt < 4; ++nt) o[nt][r] *= alpha;
      pbuf[wave][r + half * 8][m]      = f2bf(p0);
      pbuf[wave][r + half * 8][16 + m] = f2bf(p1);
    }
    // ---- re-layout P (C-layout) -> A-fragment via wave-private LDS ----
    asm volatile("s_wait_dscnt 0x0" ::: "memory");
    v16bf aP;
#pragma unroll
    for (int v = 0; v < 8; ++v) {
      int kk = 2 * (v & 3) + ((v >> 2) << 4) + half * 8;
      aP[2 * v] = pbuf[wave][m][kk];
      aP[2 * v + 1] = pbuf[wave][m][kk + 1];
    }
    // ---- O += P x V ----------------------------------------------------
#pragma unroll
    for (int nt = 0; nt < 4; ++nt) {
      v16bf bV = load_frag_bf16(vt, SEQ, head * HDIM + nt * 16, k0);
      o[nt] = __builtin_amdgcn_wmma_f32_16x16x32_bf16(false, aP, false, bV,
                                                      (short)0, o[nt], false, false);
    }
  }
  // ---- normalize + store att[s, h*64+d] (f32) -------------------------
#pragma unroll
  for (int r = 0; r < 8; ++r) {
    float inv = 1.0f / lrow[r];
    int row = qbase + r + half * 8;
#pragma unroll
    for (int nt = 0; nt < 4; ++nt)
      att[(size_t)row * QDIM + head * HDIM + nt * 16 + m] = o[nt][r] * inv;
  }
}

extern "C" void kernel_launch(void* const* d_in, const int* in_sizes, int n_in,
                              void* d_out, int out_size, void* d_ws, size_t ws_size,
                              hipStream_t stream) {
  const float* x         = (const float*)d_in[0];
  const float* wq        = (const float*)d_in[1];
  const float* w_kv_down = (const float*)d_in[2];
  const float* w_k_up    = (const float*)d_in[3];
  const float* w_v_up    = (const float*)d_in[4];
  const float* wo        = (const float*)d_in[5];

  // Workspace layout (80 MB total):
  char* ws = (char*)d_ws;
  const size_t MB = 1u << 20;
  float*  qf    = (float*)(ws + 0 * MB);    // [S,1024] f32   16 MB
  float*  kf    = (float*)(ws + 16 * MB);   // [S,1024] f32   16 MB
  float*  kvlat = (float*)(ws + 32 * MB);   // [S,512]  f32    8 MB
  __bf16* qb    = (__bf16*)(ws + 40 * MB);  // [S,1024] bf16   8 MB
  __bf16* kb    = (__bf16*)(ws + 48 * MB);  // [S,1024] bf16   8 MB
  __bf16* vt    = (__bf16*)(ws + 56 * MB);  // [1024,S] bf16   8 MB (transposed)
  float*  att   = (float*)(ws + 64 * MB);   // [S,1024] f32   16 MB

  dim3 blk(128);
  gemm_xbt<0><<<dim3(SEQ / 64, QDIM / 64), blk, 0, stream>>>(x, wq, qf, SEQ, QDIM, DMODEL);
  gemm_xbt<0><<<dim3(SEQ / 64, LATENT / 64), blk, 0, stream>>>(x, w_kv_down, kvlat, SEQ, LATENT, DMODEL);
  gemm_xbt<0><<<dim3(SEQ / 64, QDIM / 64), blk, 0, stream>>>(kvlat, w_k_up, kf, SEQ, QDIM, LATENT);
  gemm_xbt<1><<<dim3(SEQ / 64, QDIM / 64), blk, 0, stream>>>(kvlat, w_v_up, vt, SEQ, QDIM, LATENT);
  rope_cvt<<<(SEQ * NHEADS * 32 + 255) / 256, 256, 0, stream>>>(qf, kf, qb, kb);
  mla_flash<<<dim3(SEQ / 64, NHEADS), blk, 0, stream>>>(qb, kb, vt, att);
  gemm_xbt<0><<<dim3(SEQ / 64, DMODEL / 64), blk, 0, stream>>>(att, wo, (float*)d_out, SEQ, DMODEL, QDIM);
}
